// GraphWaveletPrompt_34248069218347
// MI455X (gfx1250) — compile-verified, matrix operand order checked by reading, not run on hardware
//
#include <hip/hip_runtime.h>
#include <hip/hip_bf16.h>

// ---------------------------------------------------------------------------
// GraphWaveletPrompt for MI455X (gfx1250, wave32).
//
// Math identity used: the smoothing step T(v) = 0.5*v + (0.5/norm)*scatter(v)
// is linear, so the reference's per-scale recursion collapses to
//   comb[:, s*D:(s+1)*D] = scales[s] * T^{s+1}(node_x + edge_agg)
// i.e. only S=4 scatter passes over one combined buffer Q instead of 20.
// ---------------------------------------------------------------------------

typedef __attribute__((ext_vector_type(16))) _Float16 v16h;
typedef __attribute__((ext_vector_type(8)))  float    v8f;
typedef __attribute__((ext_vector_type(4)))  _Float16 v4h;

#define DIM 128      // feature dim
#define NA  5        // anchors
#define NS  4        // scales

__device__ __forceinline__ float wave_sum32(float v) {
#pragma unroll
  for (int m = 16; m >= 1; m >>= 1) v += __shfl_xor(v, m, 32);
  return v;
}

__device__ __forceinline__ void atomAddF(float* p, float v) {
  unsafeAtomicAdd(p, v);   // lowers to global_atomic_add_f32
}

// -------------------------------------------------------------------- zero
__global__ __launch_bounds__(256) void k_zero4(float4* __restrict__ p, int n4) {
  int i = blockIdx.x * 256 + threadIdx.x;
  if (i < n4) p[i] = make_float4(0.f, 0.f, 0.f, 0.f);
}

// --------------------------------------------------- weight convert+transpose
// w1: [512,256] row-major fp32 -> w1t: [256][512] f16 (K contiguous per col)
// w2: [256,128] row-major fp32 -> w2t: [128][256] f16
__global__ __launch_bounds__(256) void k_convert_w(
    const float* __restrict__ w1, const float* __restrict__ w2,
    _Float16* __restrict__ w1t, _Float16* __restrict__ w2t) {
  int i = blockIdx.x * 256 + threadIdx.x;
  const int n1 = 512 * 256;
  const int n2 = 256 * 128;
  if (i < n1) {
    int k = i / 256, c = i % 256;
    w1t[c * 512 + k] = (_Float16)w1[i];
  } else if (i < n1 + n2) {
    int j = i - n1;
    int k = j / 128, c = j % 128;
    w2t[c * 256 + k] = (_Float16)w2[j];
  }
}

// -------------------------------------------------------------- node prompt
// Q[node] = x[node] + softmax(x @ attn_w + b) @ anchor   (one wave per node)
__global__ __launch_bounds__(256) void k_node_prompt(
    const float* __restrict__ x, const float* __restrict__ attn_w,
    const float* __restrict__ attn_b, const float* __restrict__ anchor,
    float* __restrict__ Q, int N) {
  const int wave = threadIdx.x >> 5, lane = threadIdx.x & 31;
  const int node = blockIdx.x * 8 + wave;
  if (node >= N) return;
  const float4 xs = ((const float4*)x)[(size_t)node * 32 + lane];
  const int d0 = lane * 4;
  float lg[NA];
#pragma unroll
  for (int a = 0; a < NA; ++a) {
    float p = xs.x * attn_w[(d0 + 0) * NA + a] + xs.y * attn_w[(d0 + 1) * NA + a]
            + xs.z * attn_w[(d0 + 2) * NA + a] + xs.w * attn_w[(d0 + 3) * NA + a];
    lg[a] = wave_sum32(p) + attn_b[a];
  }
  float mx = lg[0];
#pragma unroll
  for (int a = 1; a < NA; ++a) mx = fmaxf(mx, lg[a]);
  float e[NA], s = 0.f;
#pragma unroll
  for (int a = 0; a < NA; ++a) { e[a] = __expf(lg[a] - mx); s += e[a]; }
  const float inv = 1.f / s;
  float4 o = xs;
#pragma unroll
  for (int a = 0; a < NA; ++a) {
    const float w = e[a] * inv;
    o.x += w * anchor[a * DIM + d0 + 0];
    o.y += w * anchor[a * DIM + d0 + 1];
    o.z += w * anchor[a * DIM + d0 + 2];
    o.w += w * anchor[a * DIM + d0 + 3];
  }
  ((float4*)Q)[(size_t)node * 32 + lane] = o;
}

// -------------------------------------------------------------- edge prompt
// b = softmax(leaky_relu([x_src|x_dst] @ ew + eb)); ep = b @ anchor.
// Writes ep to output and atomically accumulates into Q (= edge_agg fused).
__global__ __launch_bounds__(256) void k_edge_prompt(
    const float* __restrict__ x, const int* __restrict__ src,
    const int* __restrict__ dst, const float* __restrict__ ew,
    const float* __restrict__ eb, const float* __restrict__ anchor,
    float* __restrict__ Q, float* __restrict__ ep_out, int E) {
  const int wave = threadIdx.x >> 5, lane = threadIdx.x & 31;
  const int e = blockIdx.x * 8 + wave;
  if (e >= E) return;
  const int si = src[e], di = dst[e];
  const float4 xs = ((const float4*)x)[(size_t)si * 32 + lane];
  const float4 xd = ((const float4*)x)[(size_t)di * 32 + lane];
  const int d0 = lane * 4;
  float lg[NA];
#pragma unroll
  for (int a = 0; a < NA; ++a) {
    float p = xs.x * ew[(d0 + 0) * NA + a] + xs.y * ew[(d0 + 1) * NA + a]
            + xs.z * ew[(d0 + 2) * NA + a] + xs.w * ew[(d0 + 3) * NA + a]
            + xd.x * ew[(DIM + d0 + 0) * NA + a] + xd.y * ew[(DIM + d0 + 1) * NA + a]
            + xd.z * ew[(DIM + d0 + 2) * NA + a] + xd.w * ew[(DIM + d0 + 3) * NA + a];
    float l = wave_sum32(p) + eb[a];
    lg[a] = l > 0.f ? l : 0.01f * l;   // leaky_relu(slope 0.01)
  }
  float mx = lg[0];
#pragma unroll
  for (int a = 1; a < NA; ++a) mx = fmaxf(mx, lg[a]);
  float ex[NA], s = 0.f;
#pragma unroll
  for (int a = 0; a < NA; ++a) { ex[a] = __expf(lg[a] - mx); s += ex[a]; }
  const float inv = 1.f / s;
  float4 o = make_float4(0.f, 0.f, 0.f, 0.f);
#pragma unroll
  for (int a = 0; a < NA; ++a) {
    const float w = ex[a] * inv;
    o.x += w * anchor[a * DIM + d0 + 0];
    o.y += w * anchor[a * DIM + d0 + 1];
    o.z += w * anchor[a * DIM + d0 + 2];
    o.w += w * anchor[a * DIM + d0 + 3];
  }
  ((float4*)ep_out)[(size_t)e * 32 + lane] = o;
  float* qs = Q + (size_t)si * DIM + d0;
  float* qd = Q + (size_t)di * DIM + d0;
  atomAddF(qs + 0, o.x); atomAddF(qs + 1, o.y); atomAddF(qs + 2, o.z); atomAddF(qs + 3, o.w);
  atomAddF(qd + 0, o.x); atomAddF(qd + 1, o.y); atomAddF(qd + 2, o.z); atomAddF(qd + 3, o.w);
}

// ------------------------------------------------------- wavelet scatter pass
// NB[src] += f[dst]; NB[dst] += f[src]   (one wave per edge, f32 atomics)
__global__ __launch_bounds__(256) void k_scatter(
    const float* __restrict__ f, const int* __restrict__ src,
    const int* __restrict__ dst, float* __restrict__ nb, int E) {
  const int wave = threadIdx.x >> 5, lane = threadIdx.x & 31;
  const int e = blockIdx.x * 8 + wave;
  if (e >= E) return;
  const int si = src[e], di = dst[e];
  const float4 fd = ((const float4*)f)[(size_t)di * 32 + lane];
  const float4 fs = ((const float4*)f)[(size_t)si * 32 + lane];
  const int d0 = lane * 4;
  float* ns = nb + (size_t)si * DIM + d0;
  float* nd = nb + (size_t)di * DIM + d0;
  atomAddF(ns + 0, fd.x); atomAddF(ns + 1, fd.y); atomAddF(ns + 2, fd.z); atomAddF(ns + 3, fd.w);
  atomAddF(nd + 0, fs.x); atomAddF(nd + 1, fs.y); atomAddF(nd + 2, fs.z); atomAddF(nd + 3, fs.w);
}

// -------------------------------------------------------- wavelet combine
// Q = 0.5*Q + (0.5/norm)*NB  (in place);  comb_h[:, s*D:(s+1)*D] = f16(scale*Q)
__global__ __launch_bounds__(256) void k_combine(
    float* __restrict__ Q, const float* __restrict__ nb,
    _Float16* __restrict__ combH, const float* __restrict__ scales,
    float invh, int N, int s) {
  const int i = blockIdx.x * 256 + threadIdx.x;  // over N*32 float4s
  if (i >= N * 32) return;
  float4 f = ((const float4*)Q)[i];
  const float4 b = ((const float4*)nb)[i];
  f.x = 0.5f * f.x + invh * b.x;
  f.y = 0.5f * f.y + invh * b.y;
  f.z = 0.5f * f.z + invh * b.z;
  f.w = 0.5f * f.w + invh * b.w;
  ((float4*)Q)[i] = f;
  const float sc = scales[s];
  const int n = i >> 5, l = i & 31;
  v4h h;
  h.x = (_Float16)(sc * f.x); h.y = (_Float16)(sc * f.y);
  h.z = (_Float16)(sc * f.z); h.w = (_Float16)(sc * f.w);
  *(v4h*)(&combH[(size_t)n * (NS * DIM) + s * DIM + l * 4]) = h;
}

// ------------------------------------------------------------- fused MLP
// out = relu(comb @ w1 + b1) @ w2 + b2 using v_wmma_f32_16x16x32_f16.
// Per-lane fragment index math follows CDNA5 ISA 7.12.2 (16-bit 16x32 A tile:
// lanes 0-15/16-31 hold the same M rows with K pairs interleaved by half).
__device__ __forceinline__ int kpair(int v, int half) {
  return ((v < 4) ? 2 * v : 16 + 2 * (v - 4)) + 8 * half;
}

union FragU { v16h h; unsigned u[8]; };

__global__ __launch_bounds__(256) void k_mlp(
    const _Float16* __restrict__ combH,  // [Npad, 512]
    const _Float16* __restrict__ w1t,    // [256][512] (col of w1, K contiguous)
    const _Float16* __restrict__ w2t,    // [128][256]
    const float* __restrict__ b1, const float* __restrict__ b2,
    float* __restrict__ out, int N) {
  __shared__ _Float16 Alds[16 * 512];   // 16 rows of comb (f16)  = 16 KB
  __shared__ _Float16 Hlds[16 * 256];   // hidden strip (f16)     =  8 KB
  const int tid = threadIdx.x;
  const int wave = tid >> 5, lane = tid & 31;
  const int half = lane >> 4, mn = lane & 15;
  const int row0 = blockIdx.x * 16;

  // cooperative load of A strip: 16*512 halfs = 1024 uint4
  {
    const uint4* s = (const uint4*)(combH + (size_t)row0 * 512);
    uint4* d = (uint4*)Alds;
#pragma unroll
    for (int i = 0; i < 4; ++i) d[tid + i * 256] = s[tid + i * 256];
  }
  __syncthreads();

  const v8f vzero = {0.f, 0.f, 0.f, 0.f, 0.f, 0.f, 0.f, 0.f};

  // ---- GEMM1: [16,512] x [512,256] -> Hlds. Each wave owns 32 output cols.
  v8f acc0 = vzero, acc1 = vzero;
  for (int k0 = 0; k0 < 512; k0 += 32) {
    FragU a;
#pragma unroll
    for (int v = 0; v < 8; ++v)
      a.u[v] = *(const unsigned*)(&Alds[mn * 512 + k0 + kpair(v, half)]);
    FragU bA, bB;
    const int colA = wave * 32 + mn, colB = colA + 16;
#pragma unroll
    for (int v = 0; v < 8; ++v) {
      const int kb = k0 + kpair(v, half);
      bA.u[v] = *(const unsigned*)(&w1t[colA * 512 + kb]);
      bB.u[v] = *(const unsigned*)(&w1t[colB * 512 + kb]);
    }
    acc0 = __builtin_amdgcn_wmma_f32_16x16x32_f16(false, a.h, false, bA.h,
                                                  (short)0, acc0, false, false);
    acc1 = __builtin_amdgcn_wmma_f32_16x16x32_f16(false, a.h, false, bB.h,
                                                  (short)0, acc1, false, false);
  }
  // bias + relu + f16 into Hlds (C layout: VGPR r -> M = r + 8*half, N = mn)
#pragma unroll
  for (int r = 0; r < 8; ++r) {
    const int m = r + 8 * half;
    int col = wave * 32 + mn;
    float hv = acc0[r] + b1[col];
    Hlds[m * 256 + col] = (_Float16)(hv > 0.f ? hv : 0.f);
    col += 16;
    hv = acc1[r] + b1[col];
    Hlds[m * 256 + col] = (_Float16)(hv > 0.f ? hv : 0.f);
  }
  __syncthreads();

  // ---- GEMM2: [16,256] x [256,128]. Each wave owns 16 output cols.
  v8f acc = vzero;
  for (int k0 = 0; k0 < 256; k0 += 32) {
    FragU a, b;
    const int col = wave * 16 + mn;
#pragma unroll
    for (int v = 0; v < 8; ++v) {
      const int kb = k0 + kpair(v, half);
      a.u[v] = *(const unsigned*)(&Hlds[mn * 256 + kb]);
      b.u[v] = *(const unsigned*)(&w2t[col * 256 + kb]);
    }
    acc = __builtin_amdgcn_wmma_f32_16x16x32_f16(false, a.h, false, b.h,
                                                 (short)0, acc, false, false);
  }
#pragma unroll
  for (int r = 0; r < 8; ++r) {
    const int m = r + 8 * half;
    const int col = wave * 16 + mn;
    if (row0 + m < N)
      out[(size_t)(row0 + m) * DIM + col] = acc[r] + b2[col];
  }
}

// ---------------------------------------------------------------------------
extern "C" void kernel_launch(void* const* d_in, const int* in_sizes, int n_in,
                              void* d_out, int out_size, void* d_ws, size_t ws_size,
                              hipStream_t stream) {
  const float* x           = (const float*)d_in[0];
  const int*   ei          = (const int*)d_in[1];
  const float* node_anchor = (const float*)d_in[2];
  const float* node_attn_w = (const float*)d_in[3];
  const float* node_attn_b = (const float*)d_in[4];
  const float* edge_anchor = (const float*)d_in[5];
  const float* edge_w_w    = (const float*)d_in[6];
  const float* edge_w_b    = (const float*)d_in[7];
  const float* scales      = (const float*)d_in[8];
  const float* fus_w1      = (const float*)d_in[9];
  const float* fus_b1      = (const float*)d_in[10];
  const float* fus_w2      = (const float*)d_in[11];
  const float* fus_b2      = (const float*)d_in[12];

  const int N = in_sizes[0] / DIM;
  const int E = in_sizes[1] / 2;
  const int* src = ei;
  const int* dst = ei + E;

  float* out_x  = (float*)d_out;                 // [N,128]
  float* out_ep = out_x + (size_t)N * DIM;       // [E,128]

  const int Npad = (N + 15) & ~15;
  char* ws = (char*)d_ws;
  size_t off = 0;
  float* Q = (float*)(ws + off);        off += (size_t)Npad * DIM * sizeof(float);
  float* NB = (float*)(ws + off);       off += (size_t)Npad * DIM * sizeof(float);
  _Float16* combH = (_Float16*)(ws + off); off += (size_t)Npad * NS * DIM * sizeof(_Float16);
  _Float16* w1t = (_Float16*)(ws + off);   off += (size_t)512 * 256 * sizeof(_Float16);
  _Float16* w2t = (_Float16*)(ws + off);   off += (size_t)256 * 128 * sizeof(_Float16);
  (void)off; (void)ws_size; (void)n_in; (void)out_size;

  const float norm = (float)E / (float)N + 1e-6f;
  const float invh = 0.5f / norm;

  const int n4 = N * 32;                       // float4 count per [N,128] buffer

  k_convert_w<<<(512 * 256 + 256 * 128 + 255) / 256, 256, 0, stream>>>(
      fus_w1, fus_w2, w1t, w2t);
  k_node_prompt<<<(N + 7) / 8, 256, 0, stream>>>(
      x, node_attn_w, node_attn_b, node_anchor, Q, N);
  k_edge_prompt<<<(E + 7) / 8, 256, 0, stream>>>(
      x, src, dst, edge_w_w, edge_w_b, edge_anchor, Q, out_ep, E);

  for (int s = 0; s < NS; ++s) {
    k_zero4<<<(n4 + 255) / 256, 256, 0, stream>>>((float4*)NB, n4);
    k_scatter<<<(E + 7) / 8, 256, 0, stream>>>(Q, src, dst, NB, E);
    k_combine<<<(n4 + 255) / 256, 256, 0, stream>>>(Q, NB, combH, scales,
                                                    invh, N, s);
  }

  k_mlp<<<Npad / 16, 256, 0, stream>>>(combH, w1t, w2t, fus_b1, fus_b2,
                                       out_x, N);
}
